// WordParser_52553219834677
// MI455X (gfx1250) — matrix operand order repaired and checked by reading, not compiled, not run on hardware
//
#include <hip/hip_runtime.h>
#include <hip/hip_bf16.h>

// ---------------------------------------------------------------------------
// BiLSTM dependency parser, MI455X (gfx1250, wave32, WMMA bf16 path).
// All dense math runs through v_wmma_f32_16x16x32_bf16 with fp32 accumulate.
// GEMM staging uses the CDNA5 async engine (global_load_async_to_lds_b128 +
// s_wait_asynccnt) with double-buffered LDS tiles.
// ---------------------------------------------------------------------------

typedef __attribute__((ext_vector_type(16))) __bf16 v16bf;
typedef __attribute__((ext_vector_type(8)))  __bf16 v8bf;
typedef __attribute__((ext_vector_type(8)))  float  v8f;

#define NB   32          // batch
#define NT   256         // time
#define ND   512         // hidden
#define NBT  8192        // NB*NT
#define NG   2048        // 4*ND
#define KPAD 544         // 513 padded to a multiple of 32
#define NREL 40

__device__ __forceinline__ v8f vzero() {
  v8f z = {0.f,0.f,0.f,0.f,0.f,0.f,0.f,0.f};
  return z;
}
__device__ __forceinline__ v16bf frag16(const __bf16* p0, const __bf16* p1) {
  v8bf a = *(const v8bf*)p0;
  v8bf b = *(const v8bf*)p1;
  v16bf r;
#pragma unroll
  for (int i = 0; i < 8; ++i) { r[i] = a[i]; r[i+8] = b[i]; }
  return r;
}
// A-operand fragment (16x32 bf16, row-major source). Lane owns row `row`;
// hi = lane>>4 selects the interleaved K halves per the ISA layout:
// elems 0..7 -> K = kb + hi*8 .. +7 ; elems 8..15 -> K = kb+16+hi*8 .. +7
__device__ __forceinline__ v16bf fragA(const __bf16* row, int kb, int hi) {
  return frag16(row + kb + hi*8, row + kb + 16 + hi*8);
}
// B-operand fragment (32x16 bf16) sourced from Bt[n][k] (output-col major):
// lane owns column n; elems 0..7 -> K = kb + hi*8 .. +7 ; elems 8..15 -> K = kb+16+hi*8 .. +7
__device__ __forceinline__ v16bf fragB(const __bf16* row, int kb, int hi) {
  const __bf16* p = row + kb + hi*8;
  return frag16(p, p + 16);
}
__device__ __forceinline__ v8f wmma_bf(v16bf a, v16bf b, v8f c) {
  // (neg_a, A, neg_b, B, c_mod, C, reuse_a, reuse_b)
  return __builtin_amdgcn_wmma_f32_16x16x32_bf16(false, a, false, b,
                                                 (short)0, c, false, false);
}

// --- CDNA5 async global->LDS copy (16B per lane), tracked on ASYNCcnt ------
// the LDS destination address comes from the vdst VGPR; generic pointers to __shared__ carry the
// LDS byte offset in their low 32 bits, so a trunc-cast gives the VDST value.
__device__ __forceinline__ void async_cp16(void* lds, const void* g) {
  asm volatile("global_load_async_to_lds_b128 %0, %1, off"
               :: "v"((unsigned)(unsigned long long)lds),
                  "v"((unsigned long long)g)
               : "memory");
}
__device__ __forceinline__ void wait_async_le4() {
  asm volatile("s_wait_asynccnt 4" ::: "memory");
}
__device__ __forceinline__ void wait_async_0() {
  asm volatile("s_wait_asynccnt 0" ::: "memory");
}

// ---------------------------------------------------------------------------
// Elementwise helpers: weight conversion / transpose-pad / embedding / ones
// ---------------------------------------------------------------------------
__global__ void k_cvt(const float* __restrict__ s, __bf16* __restrict__ d, long n) {
  long i = (long)blockIdx.x * blockDim.x + threadIdx.x;
  if (i < n) d[i] = (__bf16)s[i];
}

// dst[b][n][k] = (k<K0 && n<N0) ? src[b][k][n] : 0   (build Bt layout, zero pad)
__global__ void k_cvt_t_pad(const float* __restrict__ s, __bf16* __restrict__ d,
                            int K0, int N0, int KP, int NP,
                            long sstride, long dstride, long total) {
  long i = (long)blockIdx.x * blockDim.x + threadIdx.x;
  if (i >= total) return;
  int k = (int)(i % KP);
  long r = i / KP;
  int n = (int)(r % NP);
  int bb = (int)(r / NP);
  float v = (k < K0 && n < N0) ? s[bb * sstride + (long)k * N0 + n] : 0.f;
  d[bb * dstride + (long)n * KP + k] = (__bf16)v;
}

__global__ void k_embed(const int* __restrict__ words, const float* __restrict__ emb,
                        __bf16* __restrict__ x) {
  int i = blockIdx.x * blockDim.x + threadIdx.x;     // NBT*ND threads
  int row = i >> 9, dcol = i & 511;
  x[i] = (__bf16)emb[(size_t)words[row] * ND + dcol];
}

// tail of padded [NBT][KPAD] activation: col 512 = 1.0 (bias column), 513.. = 0
__global__ void k_pad_tail(__bf16* __restrict__ buf) {
  int i = blockIdx.x * blockDim.x + threadIdx.x;     // NBT*32 threads
  int row = i >> 5, c = i & 31;
  buf[(size_t)row * KPAD + 512 + c] = (c == 0) ? (__bf16)1.0f : (__bf16)0.0f;
}

// ---------------------------------------------------------------------------
// Generic tiled bf16 GEMM: C[M,N] = A[M,K] * Bt[N,K]^T.
// Preconditions (guaranteed by the host code): M, N multiples of 128,
// K multiple of 32, all operand buffers padded accordingly -> no OOB, so
// staging is pure async-to-LDS. Double-buffered: async engine fills chunk
// kc+1 while WMMAs consume chunk kc (4 async b128 per wave per chunk; async
// loads complete in order, so s_wait_asynccnt 4 == "previous chunk landed").
// ep: 0 = fp32 out + (bias1+bias2)      (LSTM pre-activation)
//     1 = bf16 out + bias1 + leakyrelu  (MLP)
//     2 = bf16 out, plain               (xd @ arc_W)
// ---------------------------------------------------------------------------
__global__ __launch_bounds__(256) void k_gemm(
    const __bf16* __restrict__ A, int lda,
    const __bf16* __restrict__ Bt, int ldb,
    int M, int N, int K,
    void* __restrict__ Cout, int ldc, int ep,
    const float* __restrict__ bias1, const float* __restrict__ bias2) {
  __shared__ __align__(16) __bf16 As[2][128][40];
  __shared__ __align__(16) __bf16 Bs[2][128][40];
  const int n0 = blockIdx.x * 128;
  const int m0 = blockIdx.y * 128;
  const int tid = threadIdx.x;
  const int w = tid >> 5, l = tid & 31;
  const int hi = l >> 4, ln = l & 15;
  const int wm = w & 3, wn = w >> 2;          // 4x2 wave grid, 32x64 per wave
  v8f acc[2][4];
#pragma unroll
  for (int i = 0; i < 2; ++i)
#pragma unroll
    for (int j = 0; j < 4; ++j) acc[i][j] = vzero();

  const int lr  = tid >> 1;                   // staging row 0..127
  const int lkh = (tid & 1) * 16;             // K half of the 32-chunk
  const int nk = K >> 5;
  const __bf16* aSrc = A + (size_t)(m0 + lr) * lda + lkh;
  const __bf16* bSrc = Bt + (size_t)(n0 + lr) * ldb + lkh;

  auto issue = [&](int kc, int buf) {
    const int k0 = kc << 5;
    async_cp16(&As[buf][lr][lkh],     aSrc + k0);
    async_cp16(&As[buf][lr][lkh + 8], aSrc + k0 + 8);
    async_cp16(&Bs[buf][lr][lkh],     bSrc + k0);
    async_cp16(&Bs[buf][lr][lkh + 8], bSrc + k0 + 8);
  };

  issue(0, 0);
  if (nk > 1) issue(1, 1);
  for (int kc = 0; kc < nk; ++kc) {
    if (kc + 1 < nk) wait_async_le4();   // chunk kc landed, kc+1 in flight
    else             wait_async_0();
    __syncthreads();                     // all waves' transfers visible
    const int cb = kc & 1;
    v16bf af0 = fragA(&As[cb][(wm * 2 + 0) * 16 + ln][0], 0, hi);
    v16bf af1 = fragA(&As[cb][(wm * 2 + 1) * 16 + ln][0], 0, hi);
#pragma unroll
    for (int nt = 0; nt < 4; ++nt) {
      v16bf bf = fragB(&Bs[cb][(wn * 4 + nt) * 16 + ln][0], 0, hi);
      acc[0][nt] = wmma_bf(af0, bf, acc[0][nt]);
      acc[1][nt] = wmma_bf(af1, bf, acc[1][nt]);
    }
    __syncthreads();                     // buffer cb free for reuse
    if (kc + 2 < nk) issue(kc + 2, cb);
  }
#pragma unroll
  for (int mt = 0; mt < 2; ++mt)
#pragma unroll
    for (int nt = 0; nt < 4; ++nt)
#pragma unroll
      for (int v = 0; v < 8; ++v) {
        const int row = m0 + (wm * 2 + mt) * 16 + v + 8 * hi;
        const int col = n0 + (wn * 4 + nt) * 16 + ln;
        if (row < M && col < N) {
          float val = acc[mt][nt][v];
          if (ep == 0) {
            val += bias1[col] + bias2[col];
            ((float*)Cout)[(size_t)row * ldc + col] = val;
          } else if (ep == 1) {
            val += bias1[col];
            val = val > 0.f ? val : 0.1f * val;
            ((__bf16*)Cout)[(size_t)row * ldc + col] = (__bf16)val;
          } else {
            ((__bf16*)Cout)[(size_t)row * ldc + col] = (__bf16)val;
          }
        }
      }
}

// ---------------------------------------------------------------------------
// Masked bidirectional LSTM scan. grid = 2 blocks (0=fwd, 1=bwd), 1024 thr
// (32 wave32 = full WGP). Per step: g = pre[t] + h @ Whh^T via WMMA, with
// gate-aligned column ownership so i/f/g/o for a given unit live in one wave:
//   wave w: batch half mt=w&1, q-tiles qt0=(w>>1)*2..+1, gates at j = g*512+q.
// h lives in LDS (bf16, A-operand layout); c and h stay in registers per
// owner lane. pre is fp32 (bias already folded by k_gemm ep=0). Whh (2 MB
// bf16) stays L2-resident across the 256 steps.
// ---------------------------------------------------------------------------
__global__ __launch_bounds__(1024) void k_lstm_scan(
    const float* __restrict__ preF, const float* __restrict__ preB,
    const __bf16* __restrict__ whhF, const __bf16* __restrict__ whhB,
    const int* __restrict__ words, __bf16* __restrict__ xout /* [NBT,1024] */) {
  __shared__ __align__(16) __bf16 hS[32][520];
  const int dir = blockIdx.x;
  const float* pre = dir ? preB : preF;
  const __bf16* whh = dir ? whhB : whhF;
  const int colofs = dir ? ND : 0;
  const int tid = threadIdx.x, w = tid >> 5, l = tid & 31;
  const int hi = l >> 4, ln = l & 15;
  const int mt = w & 1, qt0 = (w >> 1) * 2;

  for (int i = tid; i < 32 * 520; i += 1024) (&hS[0][0])[i] = (__bf16)0.0f;
  float c[2][8], hreg[2][8];
#pragma unroll
  for (int q = 0; q < 2; ++q)
#pragma unroll
    for (int v = 0; v < 8; ++v) { c[q][v] = 0.f; hreg[q][v] = 0.f; }
  __syncthreads();

  const __bf16* arow = &hS[mt * 16 + ln][0];
  for (int s = 0; s < NT; ++s) {
    const int tt = dir ? (NT - 1 - s) : s;
    v8f acc[2][4];
#pragma unroll
    for (int q = 0; q < 2; ++q)
#pragma unroll
      for (int g = 0; g < 4; ++g) acc[q][g] = vzero();

    for (int kc = 0; kc < 16; ++kc) {
      v16bf a = fragA(arow, kc * 32, hi);
#pragma unroll
      for (int q = 0; q < 2; ++q)
#pragma unroll
        for (int g = 0; g < 4; ++g) {
          const int n = g * ND + (qt0 + q) * 16 + ln;
          v16bf bb = fragB(whh + (size_t)n * ND, kc * 32, hi);
          acc[q][g] = wmma_bf(a, bb, acc[q][g]);
        }
    }
    __syncthreads();  // all hS reads done before owners overwrite
#pragma unroll
    for (int q = 0; q < 2; ++q) {
      const int qcol = (qt0 + q) * 16 + ln;
#pragma unroll
      for (int v = 0; v < 8; ++v) {
        const int bb = mt * 16 + v + 8 * hi;           // batch row
        const float* pr = pre + ((size_t)bb * NT + tt) * NG;
        float gi = acc[q][0][v] + pr[qcol];
        float gf = acc[q][1][v] + pr[ND + qcol];
        float gg = acc[q][2][v] + pr[2 * ND + qcol];
        float go = acc[q][3][v] + pr[3 * ND + qcol];
        gi = 1.f / (1.f + __expf(-gi));
        gf = 1.f / (1.f + __expf(-gf));
        gg = tanhf(gg);
        go = 1.f / (1.f + __expf(-go));
        const float cn = gf * c[q][v] + gi * gg;
        const float hn = go * tanhf(cn);
        const float m = (words[bb * NT + tt] != 0) ? 1.f : 0.f;
        c[q][v] = m * cn + (1.f - m) * c[q][v];
        hreg[q][v] = m * hn + (1.f - m) * hreg[q][v];
        hS[bb][qcol] = (__bf16)hreg[q][v];
        xout[((size_t)bb * NT + tt) * 1024 + colofs + qcol] = (__bf16)(m * hn);
      }
    }
    __syncthreads();
  }
}

// ---------------------------------------------------------------------------
// s_arc[b,x,y] = S1[b,x,:] . arcH[b,y,:]  (K=512), -inf where words[b,y]==PAD
// grid (2,2,32): 128x128 tile per block; 8 waves, wave w = rows w*16..+15.
// ---------------------------------------------------------------------------
__global__ __launch_bounds__(256) void k_arc(
    const __bf16* __restrict__ S1, const __bf16* __restrict__ H,
    const int* __restrict__ words, float* __restrict__ out) {
  const int b = blockIdx.z, x0 = blockIdx.y * 128, y0 = blockIdx.x * 128;
  const int tid = threadIdx.x, w = tid >> 5, l = tid & 31;
  const int hi = l >> 4, ln = l & 15;
  const __bf16* arow = S1 + (size_t)(b * NT + x0 + w * 16 + ln) * ND;
  v8f acc[8];
#pragma unroll
  for (int nt = 0; nt < 8; ++nt) acc[nt] = vzero();
  for (int kc = 0; kc < 16; ++kc) {
    v16bf a = fragA(arow, kc * 32, hi);
#pragma unroll
    for (int nt = 0; nt < 8; ++nt) {
      const __bf16* brow = H + (size_t)(b * NT + y0 + nt * 16 + ln) * ND;
      v16bf bb = fragB(brow, kc * 32, hi);
      acc[nt] = wmma_bf(a, bb, acc[nt]);
    }
  }
#pragma unroll
  for (int nt = 0; nt < 8; ++nt)
#pragma unroll
    for (int v = 0; v < 8; ++v) {
      const int x = x0 + w * 16 + v + 8 * hi;
      const int y = y0 + nt * 16 + ln;
      float val = acc[nt][v];
      if (words[b * NT + y] == 0) val = -__builtin_inff();
      out[((size_t)(b * NT + x)) * NT + y] = val;
    }
}

// ---------------------------------------------------------------------------
// s_rel[b,x,y,o] = (rd[b] @ relW[o]) @ rh[b]^T fused per (x-tile, o, b).
// Phase 1: U[16,544] = rd_tile @ W_o  -> bf16 LDS (17.9 KB).
// Phase 2: C[16,256] = U @ rh_b^T     -> scatter with o-stride 40.
// grid (16, 40, 32) = 20480 workgroups; K padded to 544 with zero weights.
// ---------------------------------------------------------------------------
__global__ __launch_bounds__(256) void k_rel(
    const __bf16* __restrict__ rd, const __bf16* __restrict__ rh,
    const __bf16* __restrict__ Wt /* [40][544][544] Bt layout */,
    float* __restrict__ out) {
  __shared__ __align__(16) __bf16 U[16][560];
  const int xt = blockIdx.x, o = blockIdx.y, b = blockIdx.z;
  const int tid = threadIdx.x, w = tid >> 5, l = tid & 31;
  const int hi = l >> 4, ln = l & 15;
  const __bf16* Wo = Wt + (size_t)o * (KPAD * KPAD);
  const __bf16* arow = rd + (size_t)(b * NT + xt * 16 + ln) * KPAD;

  for (int nt = w; nt < 34; nt += 8) {       // uniform per wave -> EXEC all-1s
    v8f acc = vzero();
    const __bf16* brow = Wo + (size_t)(nt * 16 + ln) * KPAD;
    for (int kc = 0; kc < 17; ++kc) {
      v16bf a = fragA(arow, kc * 32, hi);
      v16bf bb = fragB(brow, kc * 32, hi);
      acc = wmma_bf(a, bb, acc);
    }
#pragma unroll
    for (int v = 0; v < 8; ++v) U[v + 8 * hi][nt * 16 + ln] = (__bf16)acc[v];
  }
  __syncthreads();
  const __bf16* urow = &U[ln][0];
#pragma unroll
  for (int yi = 0; yi < 2; ++yi) {
    const int nt = w * 2 + yi;
    v8f acc = vzero();
    const __bf16* brow = rh + (size_t)(b * NT + nt * 16 + ln) * KPAD;
    for (int kc = 0; kc < 17; ++kc) {
      v16bf a = fragA(urow, kc * 32, hi);
      v16bf bb = fragB(brow, kc * 32, hi);
      acc = wmma_bf(a, bb, acc);
    }
#pragma unroll
    for (int v = 0; v < 8; ++v) {
      const int x = xt * 16 + v + 8 * hi;
      const int y = nt * 16 + ln;
      out[(((size_t)(b * NT + x)) * NT + y) * NREL + o] = acc[v];
    }
  }
}

// ---------------------------------------------------------------------------
// Host orchestration
// ---------------------------------------------------------------------------
extern "C" void kernel_launch(void* const* d_in, const int* in_sizes, int n_in,
                              void* d_out, int out_size, void* d_ws, size_t ws_size,
                              hipStream_t stream) {
  (void)in_sizes; (void)n_in; (void)out_size; (void)ws_size;
  const int* words = (const int*)d_in[0];
  const float* embed = (const float*)d_in[1];
  const float *fWih[3], *fWhh[3], *fbih[3], *fbhh[3];
  const float *bWih[3], *bWhh[3], *bbih[3], *bbhh[3];
  for (int lyr = 0; lyr < 3; ++lyr) {
    int base = 2 + 8 * lyr;
    fWih[lyr] = (const float*)d_in[base + 0];
    fWhh[lyr] = (const float*)d_in[base + 1];
    fbih[lyr] = (const float*)d_in[base + 2];
    fbhh[lyr] = (const float*)d_in[base + 3];
    bWih[lyr] = (const float*)d_in[base + 4];
    bWhh[lyr] = (const float*)d_in[base + 5];
    bbih[lyr] = (const float*)d_in[base + 6];
    bbhh[lyr] = (const float*)d_in[base + 7];
  }
  const float* Wmlp[4] = {(const float*)d_in[26], (const float*)d_in[28],
                          (const float*)d_in[30], (const float*)d_in[32]};
  const float* bmlp[4] = {(const float*)d_in[27], (const float*)d_in[29],
                          (const float*)d_in[31], (const float*)d_in[33]};
  const float* arcW = (const float*)d_in[34];   // [513,512]
  const float* relW = (const float*)d_in[35];   // [40,513,513]

  char* ws = (char*)d_ws;
  size_t cur = 0;
  auto alloc = [&](size_t bytes) -> char* {
    char* p = ws + cur;
    cur = (cur + bytes + 255) & ~(size_t)255;
    return p;
  };
  __bf16 *wihF[3], *wihB[3], *whhF[3], *whhB[3];
  for (int lyr = 0; lyr < 3; ++lyr) {
    size_t din = (lyr == 0) ? ND : 2 * ND;
    wihF[lyr] = (__bf16*)alloc((size_t)NG * din * 2);
    wihB[lyr] = (__bf16*)alloc((size_t)NG * din * 2);
    whhF[lyr] = (__bf16*)alloc((size_t)NG * ND * 2);
    whhB[lyr] = (__bf16*)alloc((size_t)NG * ND * 2);
  }
  __bf16* mWb[4];
  for (int i = 0; i < 4; ++i) mWb[i] = (__bf16*)alloc((size_t)ND * 1024 * 2);
  __bf16* arcWT = (__bf16*)alloc((size_t)ND * KPAD * 2);           // [512][544]
  __bf16* relWT = (__bf16*)alloc((size_t)NREL * KPAD * KPAD * 2);  // [40][544][544]
  __bf16* x0  = (__bf16*)alloc((size_t)NBT * ND * 2);
  __bf16* xc1 = (__bf16*)alloc((size_t)NBT * 1024 * 2);
  __bf16* xc2 = (__bf16*)alloc((size_t)NBT * 1024 * 2);
  float* preF = (float*)alloc((size_t)NBT * NG * 4);
  float* preB = (float*)alloc((size_t)NBT * NG * 4);
  __bf16* arcH = (__bf16*)alloc((size_t)NBT * ND * 2);
  __bf16* xd = (__bf16*)alloc((size_t)NBT * KPAD * 2);
  __bf16* rd = (__bf16*)alloc((size_t)NBT * KPAD * 2);
  __bf16* rh = (__bf16*)alloc((size_t)NBT * KPAD * 2);
  __bf16* S1 = (__bf16*)alloc((size_t)NBT * ND * 2);

  auto cvt = [&](const float* s, __bf16* d, long n) {
    k_cvt<<<(unsigned)((n + 255) / 256), 256, 0, stream>>>(s, d, n);
  };
  // weight conversions (bf16, layouts already x @ W^T friendly)
  for (int lyr = 0; lyr < 3; ++lyr) {
    long din = (lyr == 0) ? ND : 2 * ND;
    cvt(fWih[lyr], wihF[lyr], (long)NG * din);
    cvt(bWih[lyr], wihB[lyr], (long)NG * din);
    cvt(fWhh[lyr], whhF[lyr], (long)NG * ND);
    cvt(bWhh[lyr], whhB[lyr], (long)NG * ND);
  }
  for (int i = 0; i < 4; ++i) cvt(Wmlp[i], mWb[i], (long)ND * 1024);
  {  // arc_W [513,512] -> Bt [512][544] zero-padded
    long total = (long)ND * KPAD;
    k_cvt_t_pad<<<(unsigned)((total + 255) / 256), 256, 0, stream>>>(
        arcW, arcWT, 513, 512, KPAD, ND, 0, 0, total);
  }
  {  // rel_W [40,513,513] -> Bt [40][544][544] zero-padded
    long total = (long)NREL * KPAD * KPAD;
    k_cvt_t_pad<<<(unsigned)((total + 255) / 256), 256, 0, stream>>>(
        relW, relWT, 513, 513, KPAD, KPAD, (long)513 * 513, (long)KPAD * KPAD, total);
  }

  // embedding gather -> bf16
  k_embed<<<NBT * ND / 256, 256, 0, stream>>>(words, embed, x0);

  // --- 3 BiLSTM layers ---
  const __bf16* xin = x0;
  __bf16* outs[3] = {xc1, xc2, xc1};
  for (int lyr = 0; lyr < 3; ++lyr) {
    const int K = (lyr == 0) ? ND : 2 * ND;
    dim3 g(NG / 128, NBT / 128);
    k_gemm<<<g, 256, 0, stream>>>(xin, K, wihF[lyr], K, NBT, NG, K,
                                  preF, NG, 0, fbih[lyr], fbhh[lyr]);
    k_gemm<<<g, 256, 0, stream>>>(xin, K, wihB[lyr], K, NBT, NG, K,
                                  preB, NG, 0, bbih[lyr], bbhh[lyr]);
    k_lstm_scan<<<2, 1024, 0, stream>>>(preF, preB, whhF[lyr], whhB[lyr],
                                        words, outs[lyr]);
    xin = outs[lyr];
  }

  // --- 4 MLP heads (leaky relu 0.1) ---
  dim3 gm(ND / 128, NBT / 128);
  k_gemm<<<gm, 256, 0, stream>>>(xin, 1024, mWb[0], 1024, NBT, ND, 1024,
                                 arcH, ND, 1, bmlp[0], nullptr);
  k_gemm<<<gm, 256, 0, stream>>>(xin, 1024, mWb[1], 1024, NBT, ND, 1024,
                                 xd, KPAD, 1, bmlp[1], nullptr);
  k_gemm<<<gm, 256, 0, stream>>>(xin, 1024, mWb[2], 1024, NBT, ND, 1024,
                                 rh, KPAD, 1, bmlp[2], nullptr);
  k_gemm<<<gm, 256, 0, stream>>>(xin, 1024, mWb[3], 1024, NBT, ND, 1024,
                                 rd, KPAD, 1, bmlp[3], nullptr);
  k_pad_tail<<<NBT * 32 / 256, 256, 0, stream>>>(xd);
  k_pad_tail<<<NBT * 32 / 256, 256, 0, stream>>>(rh);
  k_pad_tail<<<NBT * 32 / 256, 256, 0, stream>>>(rd);

  // S1 = xd @ arc_W   (ones column folds arc_W's bias row)
  k_gemm<<<gm, 256, 0, stream>>>(xd, KPAD, arcWT, KPAD, NBT, ND, KPAD,
                                 S1, ND, 2, nullptr, nullptr);

  float* s_arc = (float*)d_out;
  float* s_rel = s_arc + (size_t)NB * NT * NT;
  k_arc<<<dim3(2, 2, NB), 256, 0, stream>>>(S1, arcH, words, s_arc);
  k_rel<<<dim3(16, NREL, NB), 256, 0, stream>>>(rd, rh, relWT, s_rel);
}
